// SparseMoE_24180665876701
// MI455X (gfx1250) — matrix operand order repaired and checked by reading, not compiled, not run on hardware
//
#include <hip/hip_runtime.h>

// ---------------- problem constants ----------------
#define BB 4
#define SS 2048
#define DD 1024
#define EE 8
#define HH 2560
#define TT (BB * SS)      // 8192 tokens
#define TOPK 2

// GEMM tiling: workgroup = 256 threads = 8 waves (2 M x 4 N), wave tile 32x32
#define MT 64
#define NT 128
#define LDK 40            // LDS row stride in bf16: 80B rows -> 16B-aligned b128 targets,
                          // 20-dword stride -> conflict-free A-fragment reads

typedef __bf16 bf16;
typedef bf16  v16bf __attribute__((ext_vector_type(16)));
typedef float v8f   __attribute__((ext_vector_type(8)));
typedef int   v4i   __attribute__((ext_vector_type(4)));

struct alignas(8) bf16x4 { bf16 x, y, z, w; };

#if defined(__gfx1250__) && __has_builtin(__builtin_amdgcn_global_load_async_to_lds_b128)
#define USE_ASYNC 1
#else
#define USE_ASYNC 0
#endif

// 16B global -> LDS copy. Async path uses GLOBAL_LOAD_ASYNC_TO_LDS_B128 (ASYNCcnt),
// fallback is a synchronous uint4 copy through VGPRs.
__device__ __forceinline__ void copy16(const void* g, void* l) {
#if USE_ASYNC
    __builtin_amdgcn_global_load_async_to_lds_b128(
        (v4i*)(uintptr_t)g, (v4i*)l, 0, 0);
#else
    *(uint4*)l = *(const uint4*)g;
#endif
}

template <int N> __device__ __forceinline__ void wait_async() {
#if USE_ASYNC
#if __has_builtin(__builtin_amdgcn_s_wait_asynccnt)
    __builtin_amdgcn_s_wait_asynccnt(N);
#else
    asm volatile("s_wait_asynccnt %0" :: "i"(N) : "memory");
#endif
#endif
}

// ---------------- utility kernels ----------------

__global__ void zero_kernel(float4* __restrict__ out4, int n4, int* __restrict__ counts) {
    int i = blockIdx.x * blockDim.x + threadIdx.x;
    if (i < n4) out4[i] = make_float4(0.f, 0.f, 0.f, 0.f);
    if (blockIdx.x == 0 && threadIdx.x < EE) counts[threadIdx.x] = 0;
}

__global__ void cvt_kernel(const float4* __restrict__ src, bf16x4* __restrict__ dst, int n4) {
    for (int i = blockIdx.x * blockDim.x + threadIdx.x; i < n4; i += gridDim.x * blockDim.x) {
        float4 f = src[i];
        bf16x4 o;
        o.x = (bf16)f.x; o.y = (bf16)f.y; o.z = (bf16)f.z; o.w = (bf16)f.w;
        dst[i] = o;
    }
}

// ---------------- router: logits, softmax, top-2, expert token lists ----------------

__global__ void router_kernel(const float* __restrict__ x, const float* __restrict__ Wr,
                              float* __restrict__ logits_out, float* __restrict__ probs_out,
                              int* __restrict__ topi_out,
                              int* __restrict__ counts, int* __restrict__ tokens,
                              float* __restrict__ gatew) {
    int wave = (blockIdx.x * blockDim.x + threadIdx.x) >> 5;  // one wave32 per token
    int lane = threadIdx.x & 31;
    if (wave >= TT) return;
    const float* xr = x + (size_t)wave * DD;

    float acc[EE];
#pragma unroll
    for (int e = 0; e < EE; e++) acc[e] = 0.f;

    for (int d = lane; d < DD; d += 32) {
        float xv = xr[d];
#pragma unroll
        for (int e = 0; e < EE; e++) acc[e] += xv * Wr[e * DD + d];
    }
#pragma unroll
    for (int e = 0; e < EE; e++) {
#pragma unroll
        for (int off = 16; off > 0; off >>= 1) acc[e] += __shfl_xor(acc[e], off, 32);
    }
    float mx = acc[0];
#pragma unroll
    for (int e = 1; e < EE; e++) mx = fmaxf(mx, acc[e]);
    float p[EE], sum = 0.f;
#pragma unroll
    for (int e = 0; e < EE; e++) { p[e] = __expf(acc[e] - mx); sum += p[e]; }
    float inv = 1.f / sum;
#pragma unroll
    for (int e = 0; e < EE; e++) p[e] *= inv;

    // top-2, ties -> lower index (matches jax.lax.top_k)
    int i1 = 0;
#pragma unroll
    for (int e = 1; e < EE; e++) if (p[e] > p[i1]) i1 = e;
    int i2 = (i1 == 0) ? 1 : 0;
#pragma unroll
    for (int e = 0; e < EE; e++) if (e != i1 && p[e] > p[i2]) i2 = e;

    if (lane == 0) {
#pragma unroll
        for (int e = 0; e < EE; e++) {
            logits_out[(size_t)wave * EE + e] = acc[e];
            probs_out[(size_t)wave * EE + e]  = p[e];
        }
        topi_out[wave * 2 + 0] = i1;
        topi_out[wave * 2 + 1] = i2;
        int s1 = atomicAdd(&counts[i1], 1);
        tokens[i1 * TT + s1] = wave;  gatew[i1 * TT + s1] = p[i1];
        int s2 = atomicAdd(&counts[i2], 1);
        tokens[i2 * TT + s2] = wave;  gatew[i2 * TT + s2] = p[i2];
    }
}

__global__ void scan_kernel(const int* __restrict__ counts, int* __restrict__ bases) {
    if (threadIdx.x == 0) {
        int b = 0;
#pragma unroll
        for (int e = 0; e < EE; e++) { bases[e] = b; b += counts[e]; }
    }
}

// ---------------- fragment helpers (wave32 WMMA layouts) ----------------

// A 16x32 bf16: lane m=l&15, hi=l>>4. dword p(0..7): k/2 = (p<4 ? hi*4+p : 8+hi*4+(p-4))
__device__ __forceinline__ v16bf load_a_frag(const bf16* rowbase, int hi) {
    const unsigned* pa = (const unsigned*)rowbase;
    union { v16bf v; unsigned u[8]; } ua;
#pragma unroll
    for (int p = 0; p < 8; p++) {
        int k2 = (p < 4) ? (hi * 4 + p) : (8 + hi * 4 + (p - 4));
        ua.u[p] = pa[k2];
    }
    return ua.v;
}

// B 32x16 bf16 (LDS as [n][k]): lane n=l&15, hi=l>>4. dword j(0..7): k/2 = (j<4 ? hi*4+j : 8+hi*4+(j-4))
__device__ __forceinline__ v16bf load_b_frag(const bf16* rowbase, int hi) {
    const unsigned* pb = (const unsigned*)rowbase;
    union { v16bf v; unsigned u[8]; } ub;
#pragma unroll
    for (int j = 0; j < 8; j++) ub.u[j] = pb[(j < 4) ? (hi * 4 + j) : (8 + hi * 4 + (j - 4))];
    return ub.v;
}

// ---------------- gate/up GEMM + SiLU fusion ----------------
// h[n_e, H] = silu(Xe @ Wg^T) * (Xe @ Wu^T)  (bf16, packed rows at bases[e])

__global__ __launch_bounds__(256)
void gateup_kernel(const bf16* __restrict__ xb,
                   const bf16* __restrict__ Wgb, const bf16* __restrict__ Wub,
                   const int* __restrict__ counts, const int* __restrict__ bases,
                   const int* __restrict__ tokens, bf16* __restrict__ hbuf) {
    int e = blockIdx.z;
    int cnt = counts[e];
    int m0 = blockIdx.x * MT;
    if (m0 >= cnt) return;                    // uniform over block
    int n0 = blockIdx.y * NT;                 // H offset
    const bf16* Wg = Wgb + (size_t)e * HH * DD;
    const bf16* Wu = Wub + (size_t)e * HH * DD;
    int base = bases[e];

    __shared__ alignas(16) bf16 sA [2][MT * LDK];
    __shared__ alignas(16) bf16 sBg[2][NT * LDK];
    __shared__ alignas(16) bf16 sBu[2][NT * LDK];
    __shared__ int sTok[MT];

    int tid  = threadIdx.x;
    int lane = tid & 31;
    int wid  = tid >> 5;
    int wm   = wid & 1;      // 0..1 (M)
    int wn   = wid >> 1;     // 0..3 (N)
    int hi   = lane >> 4;
    int l15  = lane & 15;

    if (tid < MT) {
        int s = m0 + tid;
        if (s >= cnt) s = cnt - 1;            // clamp: keep every async op issued
        sTok[tid] = tokens[e * TT + s];
    }
    __syncthreads();

    // 5 async b128 per thread per stage: 1 for A, 4 for B(gate|up)
    auto issue = [&](int buf, int kk) {
        {
            int row  = tid >> 2;
            int part = tid & 3;
            int tok  = sTok[row];
            copy16((const char*)(xb + (size_t)tok * DD + kk) + part * 16,
                   sA[buf] + row * LDK + part * 8);
        }
        {
            int rb = tid & 127;
            const bf16* Wsrc = (tid < 128) ? Wg : Wu;
            bf16* l = ((tid < 128) ? sBg[buf] : sBu[buf]) + rb * LDK;
            const char* g = (const char*)(Wsrc + (size_t)(n0 + rb) * DD + kk);
#pragma unroll
            for (int j = 0; j < 4; j++) copy16(g + j * 16, l + j * 8);
        }
    };

    v8f vz = {0.f,0.f,0.f,0.f,0.f,0.f,0.f,0.f};
    v8f accg[2][2], accu[2][2];
#pragma unroll
    for (int mi = 0; mi < 2; mi++)
#pragma unroll
        for (int ni = 0; ni < 2; ni++) { accg[mi][ni] = vz; accu[mi][ni] = vz; }

    auto compute = [&](int buf) {
        v16bf afr[2], bgf[2], buf2[2];
#pragma unroll
        for (int mi = 0; mi < 2; mi++)
            afr[mi] = load_a_frag(sA[buf] + (wm * 32 + mi * 16 + l15) * LDK, hi);
#pragma unroll
        for (int ni = 0; ni < 2; ni++) {
            int c = wn * 32 + ni * 16 + l15;
            bgf[ni]  = load_b_frag(sBg[buf] + c * LDK, hi);
            buf2[ni] = load_b_frag(sBu[buf] + c * LDK, hi);
        }
#pragma unroll
        for (int mi = 0; mi < 2; mi++)
#pragma unroll
            for (int ni = 0; ni < 2; ni++) {
                accg[mi][ni] = __builtin_amdgcn_wmma_f32_16x16x32_bf16(
                    false, afr[mi], false, bgf[ni], (short)0, accg[mi][ni], false, false);
                accu[mi][ni] = __builtin_amdgcn_wmma_f32_16x16x32_bf16(
                    false, afr[mi], false, buf2[ni], (short)0, accu[mi][ni], false, false);
            }
    };

    // async double-buffered K loop
    issue(0, 0);
    int cur = 0;
    for (int kk = 0; kk + 32 < DD; kk += 32) {
        issue(cur ^ 1, kk + 32);
        wait_async<5>();          // previous stage complete (5 new ops outstanding)
        __syncthreads();
        compute(cur);
        __syncthreads();          // all reads of 'cur' done before it is overwritten
        cur ^= 1;
    }
    wait_async<0>();
    __syncthreads();
    compute(cur);

    // epilogue: h = silu(g) * u  -> bf16
#pragma unroll
    for (int mi = 0; mi < 2; mi++) {
#pragma unroll
        for (int v = 0; v < 8; v++) {
            int srow = m0 + wm * 32 + mi * 16 + 8 * hi + v;   // slot in this expert
            if (srow < cnt) {
#pragma unroll
                for (int ni = 0; ni < 2; ni++) {
                    float g = accg[mi][ni][v];
                    float u = accu[mi][ni][v];
                    float hv = (g / (1.f + __expf(-g))) * u;
                    int col = n0 + wn * 32 + ni * 16 + l15;
                    hbuf[(size_t)(base + srow) * HH + col] = (bf16)hv;
                }
            }
        }
    }
}

// ---------------- down GEMM + scaled scatter-add ----------------

__global__ __launch_bounds__(256)
void down_kernel(const bf16* __restrict__ hbuf, const bf16* __restrict__ Wdb,
                 const int* __restrict__ counts, const int* __restrict__ bases,
                 const int* __restrict__ tokens, const float* __restrict__ gatew,
                 float* __restrict__ out) {
    int e = blockIdx.z;
    int cnt = counts[e];
    int m0 = blockIdx.x * MT;
    if (m0 >= cnt) return;
    int n0 = blockIdx.y * NT;                 // D offset
    const bf16* Wd = Wdb + (size_t)e * DD * HH;
    int base = bases[e];

    __shared__ alignas(16) bf16 sA[2][MT * LDK];
    __shared__ alignas(16) bf16 sB[2][NT * LDK];
    __shared__ int   sTok[MT];
    __shared__ float sW[MT];

    int tid  = threadIdx.x;
    int lane = tid & 31;
    int wid  = tid >> 5;
    int wm   = wid & 1;
    int wn   = wid >> 1;
    int hi   = lane >> 4;
    int l15  = lane & 15;

    if (tid < MT) {
        int s = m0 + tid;
        if (s >= cnt) s = cnt - 1;            // clamp (epilogue re-masks)
        sTok[tid] = tokens[e * TT + s];
        sW[tid]   = gatew[e * TT + s];
    }
    __syncthreads();

    // 3 async b128 per thread per stage: 1 for A, 2 for B
    auto issue = [&](int buf, int kk) {
        {
            int row  = tid >> 2;
            int part = tid & 3;
            int s    = m0 + row;
            if (s >= cnt) s = cnt - 1;
            copy16((const char*)(hbuf + (size_t)(base + s) * HH + kk) + part * 16,
                   sA[buf] + row * LDK + part * 8);
        }
        {
            int row  = tid >> 1;
            int part = tid & 1;
            const char* g = (const char*)(Wd + (size_t)(n0 + row) * HH + kk) + part * 32;
            bf16* l = sB[buf] + row * LDK + part * 16;
            copy16(g,      l);
            copy16(g + 16, l + 8);
        }
    };

    v8f vz = {0.f,0.f,0.f,0.f,0.f,0.f,0.f,0.f};
    v8f acc[2][2];
#pragma unroll
    for (int mi = 0; mi < 2; mi++)
#pragma unroll
        for (int ni = 0; ni < 2; ni++) acc[mi][ni] = vz;

    auto compute = [&](int buf) {
        v16bf afr[2], bfr[2];
#pragma unroll
        for (int mi = 0; mi < 2; mi++)
            afr[mi] = load_a_frag(sA[buf] + (wm * 32 + mi * 16 + l15) * LDK, hi);
#pragma unroll
        for (int ni = 0; ni < 2; ni++)
            bfr[ni] = load_b_frag(sB[buf] + (wn * 32 + ni * 16 + l15) * LDK, hi);
#pragma unroll
        for (int mi = 0; mi < 2; mi++)
#pragma unroll
            for (int ni = 0; ni < 2; ni++)
                acc[mi][ni] = __builtin_amdgcn_wmma_f32_16x16x32_bf16(
                    false, afr[mi], false, bfr[ni], (short)0, acc[mi][ni], false, false);
    };

    issue(0, 0);
    int cur = 0;
    for (int kk = 0; kk + 32 < HH; kk += 32) {
        issue(cur ^ 1, kk + 32);
        wait_async<3>();
        __syncthreads();
        compute(cur);
        __syncthreads();
        cur ^= 1;
    }
    wait_async<0>();
    __syncthreads();
    compute(cur);

    // epilogue: scale by gate prob, scatter-add into out
#pragma unroll
    for (int mi = 0; mi < 2; mi++) {
#pragma unroll
        for (int v = 0; v < 8; v++) {
            int lrow = wm * 32 + mi * 16 + 8 * hi + v;
            int s    = m0 + lrow;
            if (s < cnt) {
                int   tok = sTok[lrow];
                float w   = sW[lrow];
#pragma unroll
                for (int ni = 0; ni < 2; ni++) {
                    int col = n0 + wn * 32 + ni * 16 + l15;
                    atomicAdd(out + (size_t)tok * DD + col, acc[mi][ni][v] * w);
                }
            }
        }
    }
}

// ---------------- launcher ----------------

extern "C" void kernel_launch(void* const* d_in, const int* in_sizes, int n_in,
                              void* d_out, int out_size, void* d_ws, size_t ws_size,
                              hipStream_t stream) {
    const float* x  = (const float*)d_in[0];
    const float* Wr = (const float*)d_in[1];
    const float* Wg = (const float*)d_in[2];
    const float* Wu = (const float*)d_in[3];
    const float* Wd = (const float*)d_in[4];
    (void)in_sizes; (void)n_in; (void)out_size; (void)ws_size;

    // d_out layout: out [T,D] f32 | logits [T,E] f32 | probs [T,E] f32 | topi [T,2] i32
    float* out      = (float*)d_out;
    float* logits_o = out + (size_t)TT * DD;
    float* probs_o  = logits_o + (size_t)TT * EE;
    int*   topi_o   = (int*)(probs_o + (size_t)TT * EE);

    // workspace carve-up
    char* p = (char*)d_ws;
    bf16* xb   = (bf16*)p;  p += (size_t)TT * DD * 2;
    bf16* Wgb  = (bf16*)p;  p += (size_t)EE * HH * DD * 2;
    bf16* Wub  = (bf16*)p;  p += (size_t)EE * HH * DD * 2;
    bf16* Wdb  = (bf16*)p;  p += (size_t)EE * DD * HH * 2;
    bf16* hbuf = (bf16*)p;  p += (size_t)TT * TOPK * HH * 2;
    int*  counts = (int*)p;   p += 256;
    int*  bases  = (int*)p;   p += 256;
    int*  tokens = (int*)p;   p += (size_t)EE * TT * 4;
    float* gatew = (float*)p; p += (size_t)EE * TT * 4;

    {
        int n4 = (TT * DD) / 4;
        zero_kernel<<<(n4 + 255) / 256, 256, 0, stream>>>((float4*)out, n4, counts);
    }
    cvt_kernel<<<4096, 256, 0, stream>>>((const float4*)x,  (bf16x4*)xb,  (TT * DD) / 4);
    cvt_kernel<<<4096, 256, 0, stream>>>((const float4*)Wg, (bf16x4*)Wgb, (EE * HH * DD) / 4);
    cvt_kernel<<<4096, 256, 0, stream>>>((const float4*)Wu, (bf16x4*)Wub, (EE * HH * DD) / 4);
    cvt_kernel<<<4096, 256, 0, stream>>>((const float4*)Wd, (bf16x4*)Wdb, (EE * DD * HH) / 4);
    router_kernel<<<TT / 8, 256, 0, stream>>>(x, Wr, logits_o, probs_o, topi_o,
                                              counts, tokens, gatew);
    scan_kernel<<<1, 32, 0, stream>>>(counts, bases);
    gateup_kernel<<<dim3(TT / MT, HH / NT, EE), 256, 0, stream>>>(
        xb, Wgb, Wub, counts, bases, tokens, hbuf);
    down_kernel<<<dim3(TT / MT, DD / NT, EE), 256, 0, stream>>>(
        hbuf, Wdb, counts, bases, tokens, gatew, out);
}
